// SimCLRLoss_61083024883848
// MI455X (gfx1250) — matrix operand order repaired and checked by reading, not compiled, not run on hardware
//
#include <hip/hip_runtime.h>
#include <hip/hip_bf16.h>

typedef __attribute__((ext_vector_type(16))) _Float16 v16h;
typedef __attribute__((ext_vector_type(8)))  _Float16 v8h;
typedef __attribute__((ext_vector_type(4)))  _Float16 v4h;
typedef __attribute__((ext_vector_type(8)))  float    v8f;

#define NROWS 8192
#define DIM   128
#define NSEG  8                      // column segments (parallelism axis)
#define SEGW  (NROWS / NSEG)         // 1024 columns per segment
// exp(x/T) = exp2(x * log2(e)/T), T = 0.1
#define EXP_SCALE 14.426950408889634f

// ---------------------------------------------------------------------------
// Kernel 1: L2-normalize each row of z (fp32) -> f16 row-major; also zero the
// pos/neg accumulators (kernel 2 uses atomics, harness doesn't re-zero ws).
// One wave per row; lane handles 4 contiguous elements.
// ---------------------------------------------------------------------------
__global__ void simclr_normalize_kernel(const float* __restrict__ z,
                                        _Float16* __restrict__ znh,
                                        float* __restrict__ posneg) {
    const int tid = blockIdx.x * 256 + threadIdx.x;
    if (tid < 2 * NROWS) posneg[tid] = 0.f;

    const int wave = threadIdx.x >> 5;
    const int lane = threadIdx.x & 31;
    const int row  = blockIdx.x * 8 + wave;

    const float4 v = *(const float4*)(z + (size_t)row * DIM + lane * 4);
    float ss = v.x * v.x + v.y * v.y + v.z * v.z + v.w * v.w;
    #pragma unroll
    for (int off = 16; off > 0; off >>= 1)
        ss += __shfl_xor(ss, off, 32);
    const float inv = rsqrtf(ss);

    v4h h;
    h[0] = (_Float16)(v.x * inv);
    h[1] = (_Float16)(v.y * inv);
    h[2] = (_Float16)(v.z * inv);
    h[3] = (_Float16)(v.w * inv);
    *(v4h*)(znh + (size_t)row * DIM + lane * 4) = h;
}

// ---------------------------------------------------------------------------
// Kernel 2: Gram matrix via v_wmma_f32_16x16x32_f16, fused exp/mask/row-sums.
// Each wave: 32 rows (two 16-row A tiles in registers), one 1024-col segment.
// Per column tile: tacc += s (unconditional), pacc += (idx-equal) ? s : 0.
// Diagonal correction only in the <=2 tiles where j0 == i0 + t*16 (scalar
// uniform branch). pos = pacc, neg = tacc - pacc.
// C layout: lane l, VGPR v -> (m = v + 8*(l>>4), n = l&15).
// ---------------------------------------------------------------------------
__global__ void __launch_bounds__(256)
simclr_sim_kernel(const _Float16* __restrict__ znh,
                  const int* __restrict__ idx,
                  float* __restrict__ pos,
                  float* __restrict__ neg) {
    const int lane   = threadIdx.x & 31;
    const int waveId = threadIdx.x >> 5;
    const int gw     = blockIdx.x * 8 + waveId;   // 0..2047
    const int rowGrp = gw >> 3;                   // 0..255
    const int seg    = gw & (NSEG - 1);           // 0..7
    const int i0     = rowGrp * 32;
    const int j0base = seg * SEGW;

    const int laneHi = lane >> 4;     // 0: lanes 0-15, 1: lanes 16-31
    const int nloc   = lane & 15;
    const int kbA    = laneHi * 16;   // A-matrix K sub-offset per half
    const int kbB    = laneHi * 16;   // B-matrix K sub-offset per half

    // --- A fragments: two 16-row tiles, full K=128 (64 VGPRs, loop-invariant) ---
    v16h a[2][4];
    #pragma unroll
    for (int t = 0; t < 2; ++t) {
        const _Float16* rowA = znh + (size_t)(i0 + t * 16 + nloc) * DIM;
        #pragma unroll
        for (int kk = 0; kk < 4; ++kk) {
            v8h lo = *(const v8h*)(rowA + kk * 32 + kbA);
            v8h hi = *(const v8h*)(rowA + kk * 32 + kbA + 8);
            #pragma unroll
            for (int e = 0; e < 8; ++e) { a[t][kk][e] = lo[e]; a[t][kk][e + 8] = hi[e]; }
        }
    }

    // Row ids for this lane's 16 C elements: (t, v) -> i = i0 + t*16 + v + 8*laneHi
    int idxi[16];
    #pragma unroll
    for (int t = 0; t < 2; ++t)
        #pragma unroll
        for (int v = 0; v < 8; ++v)
            idxi[t * 8 + v] = idx[i0 + t * 16 + v + 8 * laneHi];

    float pacc[16], tacc[16];
    #pragma unroll
    for (int r = 0; r < 16; ++r) { pacc[r] = 0.f; tacc[r] = 0.f; }

    const int diagLane = nloc - 8 * laneHi;   // v value whose element is diagonal

    for (int jt = 0; jt < SEGW / 16; ++jt) {
        const int j0 = j0base + jt * 16;
        const _Float16* rowB = znh + (size_t)(j0 + nloc) * DIM;
        const int ij = idx[j0 + nloc];

        v8f c[2];
        c[0] = (v8f){}; c[1] = (v8f){};
        #pragma unroll
        for (int kk = 0; kk < 4; ++kk) {
            v16h b = *(const v16h*)(rowB + kk * 32 + kbB);
            c[0] = __builtin_amdgcn_wmma_f32_16x16x32_f16(
                       false, a[0][kk], false, b, (short)0, c[0], false, false);
            c[1] = __builtin_amdgcn_wmma_f32_16x16x32_f16(
                       false, a[1][kk], false, b, (short)0, c[1], false, false);
        }

        #pragma unroll
        for (int t = 0; t < 2; ++t) {
            #pragma unroll
            for (int v = 0; v < 8; ++v) {
                const float s  = __builtin_amdgcn_exp2f(c[t][v] * EXP_SCALE);
                const bool  eq = (idxi[t * 8 + v] == ij);
                pacc[t * 8 + v] += eq ? s : 0.f;   // same-id sum (incl. diag)
                tacc[t * 8 + v] += s;              // total row sum (incl. diag)
            }
            // Diagonal fix-up: scalar-uniform condition, taken in <=1 tile per t
            if (j0 == i0 + t * 16) {
                #pragma unroll
                for (int v = 0; v < 8; ++v) {
                    const float ds = (diagLane == v)
                                       ? __builtin_amdgcn_exp2f(c[t][v] * EXP_SCALE)
                                       : 0.f;
                    pacc[t * 8 + v] -= ds;
                    tacc[t * 8 + v] -= ds;
                }
            }
        }
    }

    // --- reduce across 16-lane halves (offsets < 16 stay in-half), then atomics ---
    #pragma unroll
    for (int r = 0; r < 16; ++r) {
        float p = pacc[r], tt = tacc[r];
        #pragma unroll
        for (int off = 8; off > 0; off >>= 1) {
            p  += __shfl_xor(p,  off, 32);
            tt += __shfl_xor(tt, off, 32);
        }
        if (nloc == 0) {
            const int i = i0 + (r >> 3) * 16 + (r & 7) + 8 * laneHi;
            atomicAdd(&pos[i], p);
            atomicAdd(&neg[i], tt - p);
        }
    }
}

// ---------------------------------------------------------------------------
// Kernel 3: loss = mean over rows of log((pos+neg)/pos)
// ---------------------------------------------------------------------------
__global__ void simclr_reduce_kernel(const float* __restrict__ pos,
                                     const float* __restrict__ neg,
                                     float* __restrict__ out) {
    __shared__ float red[256];
    float acc = 0.f;
    for (int r = threadIdx.x; r < NROWS; r += 256) {
        const float p = pos[r];
        acc += __logf((p + neg[r]) / p);
    }
    red[threadIdx.x] = acc;
    __syncthreads();
    #pragma unroll
    for (int s = 128; s > 0; s >>= 1) {
        if (threadIdx.x < s) red[threadIdx.x] += red[threadIdx.x + s];
        __syncthreads();
    }
    if (threadIdx.x == 0) out[0] = red[0] / (float)NROWS;
}

// ---------------------------------------------------------------------------
extern "C" void kernel_launch(void* const* d_in, const int* in_sizes, int n_in,
                              void* d_out, int out_size, void* d_ws, size_t ws_size,
                              hipStream_t stream) {
    const float* z   = (const float*)d_in[0];
    const int*   idx = (const int*)d_in[1];
    float*       out = (float*)d_out;

    // Workspace: [znh: 8192*128 f16 = 2MB][pos: 8192 f32][neg: 8192 f32]
    _Float16* znh = (_Float16*)d_ws;
    float*    pos = (float*)((char*)d_ws + (size_t)NROWS * DIM * sizeof(_Float16));
    float*    neg = pos + NROWS;

    // 1) normalize + f16 convert + zero accumulators
    simclr_normalize_kernel<<<NROWS / 8, 256, 0, stream>>>(z, znh, pos);

    // 2) WMMA Gram + fused epilogue: 2048 waves (256 row-groups x 8 segments)
    simclr_sim_kernel<<<(256 * NSEG) / 8, 256, 0, stream>>>(znh, idx, pos, neg);

    // 3) final scalar reduction
    simclr_reduce_kernel<<<1, 256, 0, stream>>>(pos, neg, out);
}
